// multihead_attention_25048249270748
// MI455X (gfx1250) — compile-verified, matrix operand order checked by reading, not run on hardware
//
#include <hip/hip_runtime.h>

typedef __bf16 bf16;
typedef __attribute__((ext_vector_type(16))) __bf16 v16bf;
typedef __attribute__((ext_vector_type(8)))  float   v8f;
typedef __attribute__((ext_vector_type(4)))  unsigned int v4u;
typedef int v4i __attribute__((__vector_size__(4 * sizeof(int))));

#define D_MODEL 1024
#define N_HEADS 16
#define D_Q     64
#define BATCH   8
#define SEQ     1024
#define NEGVAL  (-4294967295.0f)

// ---- CDNA5 async global->LDS copy (ASYNCcnt path), guarded ------------------
#if defined(__has_builtin)
#if __has_builtin(__builtin_amdgcn_global_load_async_to_lds_b128)
#define HAVE_ASYNC_LDS 1
#endif
#endif
#ifndef HAVE_ASYNC_LDS
#define HAVE_ASYNC_LDS 0
#endif

__device__ __forceinline__ void async_copy_b128(const void* g, void* l) {
#if HAVE_ASYNC_LDS
  // Param types per hipcc diagnostic: (v4i __device__*, v4i __shared__*, imm, imm).
  // generic->AS1 is identity on address bits; generic shared addr low 32 bits
  // are the LDS byte offset (aperture lives in the high bits).
  __builtin_amdgcn_global_load_async_to_lds_b128(
      (__attribute__((address_space(1))) v4i*)(size_t)g,
      (__attribute__((address_space(3))) v4i*)(unsigned int)(size_t)l, 0, 0);
#else
  *(v4u*)l = *(const v4u*)g;
#endif
}

__device__ __forceinline__ void async_wait0() {
#if HAVE_ASYNC_LDS
#if __has_builtin(__builtin_amdgcn_s_wait_asynccnt)
  __builtin_amdgcn_s_wait_asynccnt(0);
#else
  asm volatile("s_wait_asynccnt 0x0" ::: "memory");
#endif
#endif
}

// ---- WMMA fragment helpers (CDNA5 16x16x32 bf16, wave32) -------------------
// A-matrix 16x32 (ISA 7.12.2): lane m = lane&15; lanes>=16 take K+8 in each
// 16-wide K half; element i covers K = (i<8 ? i : i+8) + khi, pairs contiguous.
// B uses the symmetric layout on a transposed (n-major) tile, so one loader
// serves both.
__device__ __forceinline__ v16bf load_frag(const bf16* base, int stride,
                                           int koff, int lane) {
  const int row = lane & 15;
  const int khi = (lane >> 4) << 3;  // 0 or 8
  const bf16* p = base + row * stride + koff + khi;
  v16bf f;
#pragma unroll
  for (int i = 0; i < 8; ++i) {
    const int k = (i < 4) ? (2 * i) : (16 + 2 * (i - 4));
    f[2 * i]     = p[k];
    f[2 * i + 1] = p[k + 1];
  }
  return f;
}

__device__ __forceinline__ v8f wmma_bf16(v16bf a, v16bf b, v8f c) {
  return __builtin_amdgcn_wmma_f32_16x16x32_bf16(false, a, false, b, (short)0,
                                                 c, false, false);
}

// ---- Projection GEMM: P(bf16) = X @ W + bias  (X: 8192x1024) ---------------
// bf16 output keeps QP+KP+VP at 48 MB total -> L2-resident on MI455X (192 MB).
__global__ __launch_bounds__(256) void proj_gemm_kernel(
    const float* __restrict__ X, const float* __restrict__ W,
    const float* __restrict__ bias, bf16* __restrict__ P) {
  __shared__ __align__(16) bf16 As[128 * 32];    // A tile, row-major (m x k)
  __shared__ __align__(16) bf16 Bts[128 * 32];   // B tile transposed (n x k)

  const int t    = threadIdx.x;
  const int lane = t & 31;
  const int wave = t >> 5;
  const int wm   = wave >> 2;      // 0..1  (64-row strip)
  const int wn   = wave & 3;       // 0..3  (32-col strip)
  const int m0   = blockIdx.x * 128;
  const int n0   = blockIdx.y * 128;

  const v8f vzero = {};
  v8f acc[4][2];
#pragma unroll
  for (int i = 0; i < 4; ++i)
#pragma unroll
    for (int j = 0; j < 2; ++j) acc[i][j] = vzero;

  for (int k0 = 0; k0 < D_MODEL; k0 += 32) {
    __syncthreads();
#pragma unroll
    for (int e = 0; e < 16; ++e) {             // 128x32 A elements (f32->bf16)
      const int i = t * 16 + e;
      const int row = i >> 5, kk = i & 31;
      As[row * 32 + kk] = (bf16)X[(size_t)(m0 + row) * D_MODEL + k0 + kk];
    }
#pragma unroll
    for (int e = 0; e < 16; ++e) {             // 32x128 W elements -> Bt
      const int i = t * 16 + e;
      const int kk = i >> 7, n = i & 127;      // contiguous global reads in n
      Bts[n * 32 + kk] = (bf16)W[(size_t)(k0 + kk) * D_MODEL + n0 + n];
    }
    __syncthreads();

    v16bf af[4], bfr[2];
#pragma unroll
    for (int ms = 0; ms < 4; ++ms)
      af[ms] = load_frag(As + (wm * 64 + ms * 16) * 32, 32, 0, lane);
#pragma unroll
    for (int ns = 0; ns < 2; ++ns)
      bfr[ns] = load_frag(Bts + (wn * 32 + ns * 16) * 32, 32, 0, lane);
#pragma unroll
    for (int ms = 0; ms < 4; ++ms)
#pragma unroll
      for (int ns = 0; ns < 2; ++ns)
        acc[ms][ns] = wmma_bf16(af[ms], bfr[ns], acc[ms][ns]);
  }

  const int colL  = lane & 15;
  const int rbase = (lane >> 4) * 8;
#pragma unroll
  for (int ms = 0; ms < 4; ++ms)
#pragma unroll
    for (int ns = 0; ns < 2; ++ns) {
      const int col  = n0 + wn * 32 + ns * 16 + colL;
      const float bv = bias[col];
#pragma unroll
      for (int r = 0; r < 8; ++r) {
        const int row = m0 + wm * 64 + ms * 16 + rbase + r;
        P[(size_t)row * D_MODEL + col] = (bf16)(acc[ms][ns][r] + bv);
      }
    }
}

// ---- Per-(b,l,h) row sums of projected Q and K (for sign-masks) ------------
__global__ __launch_bounds__(256) void rowsum_kernel(
    const bf16* __restrict__ QP, const bf16* __restrict__ KP,
    float* __restrict__ qsum, float* __restrict__ ksum) {
  const int idx  = blockIdx.x * 256 + threadIdx.x;  // 0..262143
  const int half = idx >> 17;                        // 0: QP, 1: KP
  const int j    = idx & 131071;                     // (b*L + l)*H + h
  const int h    = j & 15;
  const int bl   = j >> 4;
  const bf16* p  = (half ? KP : QP) + (size_t)bl * D_MODEL + h * D_Q;
  float s = 0.0f;
#pragma unroll
  for (int i = 0; i < 64; ++i) s += (float)p[i];
  if (half) ksum[j] = s; else qsum[j] = s;
}

// ---- Flash attention: one (b,h), 128 q-rows per workgroup ------------------
__global__ __launch_bounds__(256) void attn_kernel(
    const bf16* __restrict__ QP, const bf16* __restrict__ KP,
    const bf16* __restrict__ VP, const float* __restrict__ qsum,
    const float* __restrict__ ksum, const float* __restrict__ qin,
    float* __restrict__ out) {
  __shared__ __align__(16) bf16  Qs[128 * 64];    // q-rows x d, row-major
  __shared__ __align__(16) bf16  Ks[64 * 64];     // key-rows x d  (Bt for QK^T)
  __shared__ __align__(16) bf16  Vts[64 * 64];    // d x key transposed (Bt for PV)
  __shared__ __align__(16) bf16  Pst[8][16 * 64]; // per-wave P relayout staging
  __shared__ float kmaskS[64];

  const int t     = threadIdx.x;
  const int lane  = t & 31;
  const int wave  = t >> 5;
  const int qb    = blockIdx.x;         // 0..7
  const int bh    = blockIdx.y;         // 0..127
  const int b     = bh >> 4;
  const int h     = bh & 15;
  const int qbase = qb * 128;
  const int hoff  = h * D_Q;

  // stage 128x64 Q block: layout-preserving bf16 copy -> async-to-LDS b128
  const bf16* qrow = QP + (size_t)(b * SEQ + qbase) * D_MODEL + hoff;
#pragma unroll
  for (int e = 0; e < 4; ++e) {
    const int i = e * 256 + t;          // 1024 chunks of 16B
    const int qr = i >> 3, part = i & 7;
    async_copy_b128(qrow + (size_t)qr * D_MODEL + part * 8,
                    &Qs[qr * 64 + part * 8]);
  }
  async_wait0();
  __syncthreads();

  const v16bf qa0 = load_frag(Qs + wave * 16 * 64, 64, 0, lane);
  const v16bf qa1 = load_frag(Qs + wave * 16 * 64, 64, 32, lane);

  const v8f vzero = {};
  v8f o[4];
  float mrow[8], lrow[8];
#pragma unroll
  for (int d = 0; d < 4; ++d) o[d] = vzero;
#pragma unroll
  for (int r = 0; r < 8; ++r) { mrow[r] = -3.0e38f; lrow[r] = 0.0f; }

  const int colL  = lane & 15;
  const int rbase = (lane >> 4) * 8;
  const int nkb   = (qbase + 128) >> 6;   // causal: skip fully-masked blocks

  for (int kb = 0; kb < nkb; ++kb) {
    const int kbase = kb * 64;
    __syncthreads();                      // prev-iteration Ks/Vts consumers done

    // K tile (row-major, layout-preserving) via async-to-LDS
    const bf16* krow = KP + (size_t)(b * SEQ + kbase) * D_MODEL + hoff;
#pragma unroll
    for (int e = 0; e < 2; ++e) {
      const int i = e * 256 + t;          // 512 chunks of 16B
      const int kk = i >> 3, part = i & 7;
      async_copy_b128(krow + (size_t)kk * D_MODEL + part * 8,
                      &Ks[kk * 64 + part * 8]);
    }
    // V tile transposed (d-major) -> element scatter
    const bf16* vrow = VP + (size_t)(b * SEQ + kbase) * D_MODEL + hoff;
#pragma unroll
    for (int e = 0; e < 16; ++e) {
      const int i = t * 16 + e;
      const int kk = i >> 6, d = i & 63;
      Vts[d * 64 + kk] = vrow[(size_t)kk * D_MODEL + d];
    }
    if (t < 64) kmaskS[t] = ksum[(size_t)(b * SEQ + kbase + t) * N_HEADS + h];
    async_wait0();
    __syncthreads();

    // S = (Q K^T)/8 for this wave's 16 rows x 64 keys (4 N-subtiles)
    v8f s[4];
#pragma unroll
    for (int ns = 0; ns < 4; ++ns) {
      const v16bf kf0 = load_frag(Ks + ns * 16 * 64, 64, 0, lane);
      const v16bf kf1 = load_frag(Ks + ns * 16 * 64, 64, 32, lane);
      v8f c = vzero;
      c = wmma_bf16(qa0, kf0, c);
      c = wmma_bf16(qa1, kf1, c);
      s[ns] = c;
    }

    // scale + key/causal masks; per-row block max
    float bmax[8];
#pragma unroll
    for (int r = 0; r < 8; ++r) bmax[r] = -3.0e38f;
#pragma unroll
    for (int ns = 0; ns < 4; ++ns) {
      const int  kg  = kbase + ns * 16 + colL;
      const bool kmz = (kmaskS[ns * 16 + colL] == 0.0f);
#pragma unroll
      for (int r = 0; r < 8; ++r) {
        const int qg = qbase + wave * 16 + rbase + r;
        float v = s[ns][r] * 0.125f;                 // 1/sqrt(64)
        if (kmz || kg > qg) v = NEGVAL;
        s[ns][r] = v;
        bmax[r]  = fmaxf(bmax[r], v);
      }
    }
    // online-softmax stats; rows live in 16-lane halves of the C layout
#pragma unroll
    for (int r = 0; r < 8; ++r) {
#pragma unroll
      for (int off = 1; off < 16; off <<= 1)
        bmax[r] = fmaxf(bmax[r], __shfl_xor(bmax[r], off, 16));
      const float mnew  = fmaxf(mrow[r], bmax[r]);
      const float alpha = __expf(mrow[r] - mnew);
      mrow[r] = mnew;
      lrow[r] *= alpha;
#pragma unroll
      for (int d = 0; d < 4; ++d) o[d][r] *= alpha;
    }

    float rs[8];
#pragma unroll
    for (int r = 0; r < 8; ++r) rs[r] = 0.0f;
#pragma unroll
    for (int ns = 0; ns < 4; ++ns)
#pragma unroll
      for (int r = 0; r < 8; ++r) {
        const float p = __expf(s[ns][r] - mrow[r]);
        s[ns][r] = p;
        rs[r]   += p;
      }
#pragma unroll
    for (int r = 0; r < 8; ++r) {
#pragma unroll
      for (int off = 1; off < 16; off <<= 1)
        rs[r] += __shfl_xor(rs[r], off, 16);
      lrow[r] += rs[r];
    }

    // C-layout P -> row-major bf16 staging -> A-fragments (wave-private)
    bf16* ps = &Pst[wave][0];
#pragma unroll
    for (int ns = 0; ns < 4; ++ns)
#pragma unroll
      for (int r = 0; r < 8; ++r)
        ps[(rbase + r) * 64 + ns * 16 + colL] = (bf16)s[ns][r];
    asm volatile("s_wait_dscnt 0x0" ::: "memory");   // same-wave LDS RAW fence

    const v16bf pa0 = load_frag(ps, 64, 0, lane);
    const v16bf pa1 = load_frag(ps, 64, 32, lane);
#pragma unroll
    for (int dsb = 0; dsb < 4; ++dsb) {
      const v16bf vf0 = load_frag(Vts + dsb * 16 * 64, 64, 0, lane);
      const v16bf vf1 = load_frag(Vts + dsb * 16 * 64, 64, 32, lane);
      o[dsb] = wmma_bf16(pa0, vf0, o[dsb]);
      o[dsb] = wmma_bf16(pa1, vf1, o[dsb]);
    }
  }

  // epilogue: 1/l normalize, query mask, residual add
#pragma unroll
  for (int r = 0; r < 8; ++r) {
    const int   qg = qbase + wave * 16 + rbase + r;
    const float qm = qsum[(size_t)(b * SEQ + qg) * N_HEADS + h];
    const float fac = (qm == 0.0f) ? 0.0f : (1.0f / lrow[r]);
#pragma unroll
    for (int dsb = 0; dsb < 4; ++dsb) {
      const size_t oi =
          (size_t)(b * SEQ + qg) * D_MODEL + hoff + dsb * 16 + colL;
      out[oi] = o[dsb][r] * fac + qin[oi];
    }
  }
}

// ---- launch ----------------------------------------------------------------
extern "C" void kernel_launch(void* const* d_in, const int* in_sizes, int n_in,
                              void* d_out, int out_size, void* d_ws,
                              size_t ws_size, hipStream_t stream) {
  const float* q  = (const float*)d_in[0];
  const float* k  = (const float*)d_in[1];
  // d_in[2] = subseq_mask (exact causal triu; applied analytically)
  const float* Wq = (const float*)d_in[3];
  const float* bq = (const float*)d_in[4];
  const float* Wk = (const float*)d_in[5];
  const float* bk = (const float*)d_in[6];
  const float* Wv = (const float*)d_in[7];
  const float* bv = (const float*)d_in[8];
  float* out = (float*)d_out;

  const size_t PROJ = (size_t)BATCH * SEQ * D_MODEL;   // 8M elements each
  bf16* QPb = (bf16*)d_ws;
  bf16* KPb = QPb + PROJ;
  bf16* VPb = KPb + PROJ;
  float* qs = (float*)(VPb + PROJ);
  float* ks = qs + (size_t)BATCH * SEQ * N_HEADS;

  dim3 pg(64, 8);  // 8192/128 x 1024/128
  proj_gemm_kernel<<<pg, 256, 0, stream>>>(q, Wq, bq, QPb);
  proj_gemm_kernel<<<pg, 256, 0, stream>>>(k, Wk, bk, KPb);
  proj_gemm_kernel<<<pg, 256, 0, stream>>>(k, Wv, bv, VPb);
  rowsum_kernel<<<1024, 256, 0, stream>>>(QPb, KPb, qs, ks);
  attn_kernel<<<dim3(8, 128), 256, 0, stream>>>(QPb, KPb, VPb, qs, ks, q, out);

  (void)in_sizes; (void)n_in; (void)out_size; (void)ws_size;
}
